// HGAT_49211735278206
// MI455X (gfx1250) — matrix operand order repaired and checked by reading, not compiled, not run on hardware
//
#include <hip/hip_runtime.h>

#define T_ 3
#define N_ 2048
#define F_ 300
#define D_ 512
#define H_ 50
#define C_ 18

typedef __bf16 bf16_t;
typedef __attribute__((ext_vector_type(16))) __bf16 v16bf;
typedef __attribute__((ext_vector_type(8)))  __bf16 v8bf;
typedef __attribute__((ext_vector_type(2)))  __bf16 v2bf;
typedef __attribute__((ext_vector_type(8)))  float  v8f;
typedef __attribute__((ext_vector_type(4)))  float  v4f;

static __device__ __forceinline__ v8f zero8() {
  v8f z = {0.f,0.f,0.f,0.f,0.f,0.f,0.f,0.f};
  return z;
}

static __device__ __forceinline__ v8f wmma_bf16(v16bf a, v16bf b, v8f c) {
  // (neg_a, A, neg_b, B, c_mod, C, reuse_a, reuse_b)
  return __builtin_amdgcn_wmma_f32_16x16x32_bf16(false, a, false, b, (short)0, c, false, false);
}

// A fragment (16x32 bf16), row-major tile: caller passes &tile[row*ld + kc].
// elements 0..7 = k in [8*hf, 8*hf+8), elements 8..15 = k in [16+8*hf, 24+8*hf)
static __device__ __forceinline__ v16bf frag_a_bf16(const bf16_t* row_kc, int hf) {
  const bf16_t* p = row_kc + 8*hf;
  v8bf lo = *(const v8bf*)p;
  v8bf hi = *(const v8bf*)(p + 16);
  return __builtin_shufflevector(lo, hi, 0,1,2,3,4,5,6,7,8,9,10,11,12,13,14,15);
}

// Same A layout but converting from f32 source (used for adj, exact 0/1 in bf16)
static __device__ __forceinline__ v16bf frag_a_f32(const float* row_kc, int hf) {
  const float* p = row_kc + 8*hf;
  v4f a0 = *(const v4f*)(p);
  v4f a1 = *(const v4f*)(p + 4);
  v4f a2 = *(const v4f*)(p + 16);
  v4f a3 = *(const v4f*)(p + 20);
  v16bf r;
#pragma unroll
  for (int i = 0; i < 4; i++) {
    r[i]    = (bf16_t)a0[i];
    r[4+i]  = (bf16_t)a1[i];
    r[8+i]  = (bf16_t)a2[i];
    r[12+i] = (bf16_t)a3[i];
  }
  return r;
}

// B fragment (32x16 bf16) from B^T layout [col][k]: caller passes &bt[col*ld + kc].
// elements 0..7 = k in [8*hf, 8*hf+8), elements 8..15 = k in [16+8*hf, 24+8*hf)
static __device__ __forceinline__ v16bf frag_b_bf16(const bf16_t* col_kc, int hf) {
  const bf16_t* p = col_kc + 8*hf;
  v8bf lo = *(const v8bf*)p;
  v8bf hi = *(const v8bf*)(p + 16);
  return __builtin_shufflevector(lo, hi, 0,1,2,3,4,5,6,7,8,9,10,11,12,13,14,15);
}

static __device__ __forceinline__ void cvt_store_pair(bf16_t* dst, float a, float b) {
  v2bf p;
  p[0] = (bf16_t)a;
  p[1] = (bf16_t)b;
  *(v2bf*)dst = p;
}

// Stage a full 16x32 f32 tile (row stride ld) into a bf16 LDS tile, paired loads.
template <int BLOCK>
static __device__ __forceinline__ void stage_a_16x32(bf16_t* lds, const float* src,
                                                     size_t ld, int tid) {
#pragma unroll
  for (int base = 0; base < 512; base += BLOCK * 2) {
    int idx = base + tid * 2;
    int r = idx >> 5, kk = idx & 31;
    const float2 f = *(const float2*)&src[(size_t)r * ld + kk];
    cvt_store_pair(&lds[idx], f.x, f.y);
  }
}

static __device__ __forceinline__ float wred_sum(float v) {
#pragma unroll
  for (int o = 16; o > 0; o >>= 1) v += __shfl_xor(v, o, 32);
  return v;
}
static __device__ __forceinline__ float wred_max(float v) {
#pragma unroll
  for (int o = 16; o > 0; o >>= 1) v = fmaxf(v, __shfl_xor(v, o, 32));
  return v;
}
static __device__ __forceinline__ float hred16_sum(float v) {
#pragma unroll
  for (int o = 8; o > 0; o >>= 1) v += __shfl_xor(v, o, 16);
  return v;
}
static __device__ __forceinline__ float hred16_max(float v) {
#pragma unroll
  for (int o = 8; o > 0; o >>= 1) v = fmaxf(v, __shfl_xor(v, o, 16));
  return v;
}
static __device__ __forceinline__ float leakyf(float v, float s) {
  return v >= 0.f ? v : s * v;
}

// ---------------------------------------------------------------------------
// Prep: transpose + pad + bf16-convert the small weights:
//   at1_W [T,D,H] -> at1WT [T,64,512] bf16 (h>=50 zero)
//   W2    [D,C]   -> W2T   [32,512]   bf16 (c>=18 zero)
// ---------------------------------------------------------------------------
__global__ __launch_bounds__(256) void prep_kernel(const float* __restrict__ at1W,
                                                   const float* __restrict__ W2,
                                                   bf16_t* __restrict__ at1WT,
                                                   bf16_t* __restrict__ W2T) {
  int i = blockIdx.x * 256 + threadIdx.x;
  const int n1 = T_ * 64 * D_;
  if (i < n1) {
    int k = i / (64 * D_);
    int rem = i % (64 * D_);
    int hcol = rem / D_;
    int d = rem % D_;
    float v = (hcol < H_) ? at1W[((size_t)k * D_ + d) * H_ + hcol] : 0.f;
    at1WT[i] = (bf16_t)v;
  } else {
    int j = i - n1;
    if (j < 32 * D_) {
      int c = j / D_, d = j % D_;
      float v = (c < C_) ? W2[(size_t)d * C_ + c] : 0.f;
      W2T[j] = (bf16_t)v;
    }
  }
}

// ---------------------------------------------------------------------------
// h[k] = x[k] @ W1[k]   (2048x300 @ 300x512), K padded 300->320.
// 8 waves: 16-row tile x 512 cols (4 col-tiles per wave). A and B^T staged in LDS.
// First 9 K-chunks are fully in-bounds (vectorized, branch-free staging);
// only the 10th chunk carries the 300->320 zero-pad predicate.
// Writes h (f32, [T,N,D]) and hT (bf16, [T,D,N]) for the attention-GEMM B operand.
// ---------------------------------------------------------------------------
__global__ __launch_bounds__(256) void proj_kernel(const float* __restrict__ x,
                                                   const float* __restrict__ W1,
                                                   float* __restrict__ hout,
                                                   bf16_t* __restrict__ hT) {
  __shared__ __attribute__((aligned(32))) bf16_t ldsA[16 * 32];
  __shared__ __attribute__((aligned(32))) bf16_t ldsB[512 * 32];
  const int k = blockIdx.x;
  const int nb = blockIdx.y * 16;
  const int tid = threadIdx.x;
  const int lane = tid & 31, wave = tid >> 5;
  const int l16 = lane & 15, hf = lane >> 4;
  const float* xk = x + (size_t)k * N_ * F_;
  const float* Wk = W1 + (size_t)k * F_ * D_;
  v8f acc[4];
#pragma unroll
  for (int j = 0; j < 4; j++) acc[j] = zero8();

  for (int chunk = 0; chunk < 10; chunk++) {
    const int kc = chunk * 32;
    if (chunk < 9) {
      // full chunk: vectorized staging, no bounds checks
      stage_a_16x32<256>(ldsA, xk + (size_t)nb * F_ + kc, F_, tid);
      for (int i4 = tid * 4; i4 < 32 * 512; i4 += 256 * 4) {
        int kk = i4 >> 9, d = i4 & 511;  // d multiple of 4
        const float4 f = *(const float4*)&Wk[(size_t)(kc + kk) * D_ + d];
        ldsB[(d + 0) * 32 + kk] = (bf16_t)f.x;
        ldsB[(d + 1) * 32 + kk] = (bf16_t)f.y;
        ldsB[(d + 2) * 32 + kk] = (bf16_t)f.z;
        ldsB[(d + 3) * 32 + kk] = (bf16_t)f.w;
      }
    } else {
      // tail chunk kc=288: kk < 12 valid, rest zero
      for (int i = tid; i < 16 * 32; i += 256) {
        int r = i >> 5, kk = i & 31;
        float f = (kk < F_ - 288) ? xk[(size_t)(nb + r) * F_ + 288 + kk] : 0.f;
        ldsA[i] = (bf16_t)f;
      }
      for (int i = tid; i < 32 * 512; i += 256) {
        int kk = i >> 9, d = i & 511;
        float f = (kk < F_ - 288) ? Wk[(size_t)(288 + kk) * D_ + d] : 0.f;
        ldsB[d * 32 + kk] = (bf16_t)f;
      }
    }
    __syncthreads();
    v16bf a = frag_a_bf16(&ldsA[l16 * 32], hf);
#pragma unroll
    for (int j = 0; j < 4; j++) {
      int dt = wave * 4 + j;
      v16bf b = frag_b_bf16(&ldsB[(dt * 16 + l16) * 32], hf);
      acc[j] = wmma_bf16(a, b, acc[j]);
    }
    __syncthreads();
  }
#pragma unroll
  for (int j = 0; j < 4; j++) {
    int d = (wave * 4 + j) * 16 + l16;
    v8bf pk;
#pragma unroll
    for (int v = 0; v < 8; v++) {
      int n = nb + v + 8 * hf;
      hout[((size_t)k * N_ + n) * D_ + d] = acc[j][v];
      pk[v] = (bf16_t)acc[j][v];
    }
    *(v8bf*)&hT[((size_t)k * D_ + d) * (size_t)N_ + nb + 8 * hf] = pk;
  }
}

// ---------------------------------------------------------------------------
// s1[k,n] = h[k,n,:]·a1[k]   s2[k,t,m] = h[t,m,:]·a2[k]   (wave-per-row reduce)
// ---------------------------------------------------------------------------
__global__ __launch_bounds__(256) void s12_kernel(const float* __restrict__ h,
                                                  const float* __restrict__ a1,
                                                  const float* __restrict__ a2,
                                                  float* __restrict__ s1,
                                                  float* __restrict__ s2) {
  int wid = blockIdx.x * 8 + (threadIdx.x >> 5);
  int lane = threadIdx.x & 31;
  if (wid < T_ * N_) {
    int k = wid / N_, n = wid % N_;
    const float* hp = h + ((size_t)k * N_ + n) * D_;
    const float* ap = a1 + k * D_;
    float p = 0.f;
    for (int d = lane; d < D_; d += 32) p += hp[d] * ap[d];
    p = wred_sum(p);
    if (lane == 0) s1[wid] = p;
  } else {
    int idx = wid - T_ * N_;
    int kt = idx / N_, m = idx % N_;
    int k = kt / T_, t = kt % T_;
    const float* hp = h + ((size_t)t * N_ + m) * D_;
    const float* ap = a2 + k * D_;
    float p = 0.f;
    for (int d = lane; d < D_; d += 32) p += hp[d] * ap[d];
    p = wred_sum(p);
    if (lane == 0) s2[idx] = p;
  }
}

// ---------------------------------------------------------------------------
// Fused node attention per (k,t) pair, 16-row tile:
//   pass1: rowmax over masked leaky(s1+s2), deg = sum adj
//   pass2: softmax denominator
//   pass3: att = p*gamma*deg/sum + 0.9*adj  ->  bf16 LDS tile (16x2048 = 64KB)
//   then  z1[k,t,rows,:] = att_tile @ h[t] + b1  via WMMA over K=2048
// att tensor (151MB) never touches HBM.
// ---------------------------------------------------------------------------
__global__ __launch_bounds__(256) void att_kernel(const float* __restrict__ adj,
                                                  const float* __restrict__ s1,
                                                  const float* __restrict__ s2,
                                                  const bf16_t* __restrict__ hT,
                                                  const float* __restrict__ b1,
                                                  float* __restrict__ z1) {
  __shared__ __attribute__((aligned(32))) bf16_t attb[16 * N_];  // 64KB
  const int p = blockIdx.x;          // k*T + t
  const int k = p / T_, t = p % T_;
  const int nb = blockIdx.y * 16;
  const int tid = threadIdx.x;
  const int r = tid >> 4, c16 = tid & 15;
  const float* adjrow = adj + ((size_t)p * N_ + nb + r) * N_;
  const float s1v = s1[k * N_ + nb + r];
  const float* s2p = s2 + (size_t)p * N_;

  // pass 1: row max of masked e, and degree
  float mx = -3.0e38f, dg = 0.f;
  for (int m = c16; m < N_; m += 16) {
    float a = adjrow[m];
    if (a > 0.f) {
      mx = fmaxf(mx, leakyf(s1v + s2p[m], 0.2f));
      dg += a;
    }
  }
  mx = hred16_max(mx);
  dg = hred16_sum(dg);
  // pass 2: softmax denominator
  float sm = 0.f;
  for (int m = c16; m < N_; m += 16) {
    float a = adjrow[m];
    if (a > 0.f) sm += __expf(leakyf(s1v + s2p[m], 0.2f) - mx);
  }
  sm = hred16_sum(sm);
  const float scale = (sm > 0.f) ? (0.1f * dg / sm) : 0.f;
  // pass 3: final mixed attention weights -> bf16 LDS tile
  for (int m = c16; m < N_; m += 16) {
    float a = adjrow[m];
    float val = 0.f;
    if (a > 0.f)
      val = __expf(leakyf(s1v + s2p[m], 0.2f) - mx) * scale + 0.9f * a;
    attb[r * N_ + m] = (bf16_t)val;
  }
  __syncthreads();

  // GEMM: [16 x 2048] @ [2048 x 512]; 8 waves x 4 col-tiles
  const int lane = tid & 31, wave = tid >> 5;
  const int l16 = lane & 15, hf = lane >> 4;
  const bf16_t* hTt = hT + (size_t)t * D_ * N_;
  v8f acc[4];
#pragma unroll
  for (int j = 0; j < 4; j++) acc[j] = zero8();
  for (int kc = 0; kc < N_; kc += 32) {
    if (kc + 256 < N_) {
#pragma unroll
      for (int j = 0; j < 4; j++) {
        int d = (wave * 4 + j) * 16 + l16;
        __builtin_prefetch(&hTt[(size_t)d * N_ + kc + 256], 0, 1);
      }
    }
    v16bf a = frag_a_bf16(&attb[l16 * N_ + kc], hf);
#pragma unroll
    for (int j = 0; j < 4; j++) {
      int d = (wave * 4 + j) * 16 + l16;
      v16bf b = frag_b_bf16(&hTt[(size_t)d * N_ + kc], hf);
      acc[j] = wmma_bf16(a, b, acc[j]);
    }
  }
#pragma unroll
  for (int j = 0; j < 4; j++) {
    int d = (wave * 4 + j) * 16 + l16;
    float bb = b1[d];
#pragma unroll
    for (int v = 0; v < 8; v++) {
      int n = nb + v + 8 * hf;
      z1[((size_t)p * N_ + n) * D_ + d] = acc[j][v] + bb;
    }
  }
}

// ---------------------------------------------------------------------------
// xl[k,t,n,h] = z1[k,t,n,:] @ at1_W[k] + at1_b[k]   (K=512, H padded to 64)
// 4 waves, one 16-wide h-tile each; A staged in LDS.
// ---------------------------------------------------------------------------
__global__ __launch_bounds__(128) void xl_kernel(const float* __restrict__ z1,
                                                 const bf16_t* __restrict__ at1WT,
                                                 const float* __restrict__ at1b,
                                                 float* __restrict__ xl) {
  __shared__ __attribute__((aligned(32))) bf16_t ldsA[16 * 32];
  const int p = blockIdx.x;
  const int k = p / T_;
  const int nb = blockIdx.y * 16;
  const int tid = threadIdx.x;
  const int lane = tid & 31, wave = tid >> 5;
  const int l16 = lane & 15, hf = lane >> 4;
  const float* zp = z1 + ((size_t)p * N_ + nb) * D_;
  const bf16_t* Wp = at1WT + (size_t)k * 64 * D_;
  v8f acc = zero8();
  for (int kc = 0; kc < D_; kc += 32) {
    stage_a_16x32<128>(ldsA, zp + kc, D_, tid);
    __syncthreads();
    v16bf a = frag_a_bf16(&ldsA[l16 * 32], hf);
    int hcol = wave * 16 + l16;
    v16bf b = frag_b_bf16(&Wp[(size_t)hcol * D_ + kc], hf);
    acc = wmma_bf16(a, b, acc);
    __syncthreads();
  }
  int hcol = wave * 16 + l16;
  if (hcol < H_) {
    float bb = at1b[k * H_ + hcol];
#pragma unroll
    for (int v = 0; v < 8; v++) {
      int n = nb + v + 8 * hf;
      xl[((size_t)p * N_ + n) * H_ + hcol] = acc[v] + bb;
    }
  }
}

// ---------------------------------------------------------------------------
// Self-attention 1 (softmax over type axis, 3 entries) + *3 + ReLU -> x1[T,N,D]
// wave per (k,n)
// ---------------------------------------------------------------------------
__global__ __launch_bounds__(256) void selfatt1_kernel(const float* __restrict__ xl,
                                                       const float* __restrict__ at1a,
                                                       const float* __restrict__ z1,
                                                       float* __restrict__ x1) {
  const int k = blockIdx.x;
  const int wave = threadIdx.x >> 5, lane = threadIdx.x & 31;
  const int n = blockIdx.y * 8 + wave;
  const float* av = at1a + k * 2 * H_;
  float part[T_];
  float pself = 0.f;
#pragma unroll
  for (int t = 0; t < T_; t++) {
    const float* xp = xl + ((size_t)(k * T_ + t) * N_ + n) * H_;
    float pt = 0.f;
    for (int hh = lane; hh < H_; hh += 32) {
      float v = xp[hh];
      pt += v * av[hh];
      if (t == k) pself += v * av[H_ + hh];
    }
    part[t] = wred_sum(pt);
  }
  pself = wred_sum(pself);
  float U0 = leakyf(part[0] + pself, 0.01f);
  float U1 = leakyf(part[1] + pself, 0.01f);
  float U2 = leakyf(part[2] + pself, 0.01f);
  float um = fmaxf(U0, fmaxf(U1, U2));
  float e0 = __expf(U0 - um), e1 = __expf(U1 - um), e2 = __expf(U2 - um);
  float inv = 1.f / (e0 + e1 + e2);
  float w0 = e0 * inv, w1 = e1 * inv, w2 = e2 * inv;
  const float* zp0 = z1 + ((size_t)(k * T_ + 0) * N_ + n) * D_;
  const float* zp1 = z1 + ((size_t)(k * T_ + 1) * N_ + n) * D_;
  const float* zp2 = z1 + ((size_t)(k * T_ + 2) * N_ + n) * D_;
  float* xo = x1 + ((size_t)k * N_ + n) * D_;
  for (int d = lane; d < D_; d += 32) {
    float v = 3.f * (w0 * zp0[d] + w1 * zp1[d] + w2 * zp2[d]);
    xo[d] = fmaxf(v, 0.f);
  }
}

// ---------------------------------------------------------------------------
// sup = x1 @ W2 (C padded to 32), stored transposed bf16: supT[T,32,N]
// 2 waves, one 16-wide c-tile each
// ---------------------------------------------------------------------------
__global__ __launch_bounds__(64) void sup_kernel(const float* __restrict__ x1,
                                                 const bf16_t* __restrict__ W2T,
                                                 bf16_t* __restrict__ supT) {
  __shared__ __attribute__((aligned(32))) bf16_t ldsA[16 * 32];
  const int t = blockIdx.x;
  const int nb = blockIdx.y * 16;
  const int tid = threadIdx.x;
  const int lane = tid & 31, wave = tid >> 5;
  const int l16 = lane & 15, hf = lane >> 4;
  const float* xp = x1 + ((size_t)t * N_ + nb) * D_;
  v8f acc = zero8();
  for (int kc = 0; kc < D_; kc += 32) {
    stage_a_16x32<64>(ldsA, xp + kc, D_, tid);
    __syncthreads();
    v16bf a = frag_a_bf16(&ldsA[l16 * 32], hf);
    int c = wave * 16 + l16;
    v16bf b = frag_b_bf16(&W2T[(size_t)c * D_ + kc], hf);
    acc = wmma_bf16(a, b, acc);
    __syncthreads();
  }
  int c = wave * 16 + l16;
  v8bf pk;
#pragma unroll
  for (int v = 0; v < 8; v++) pk[v] = (bf16_t)acc[v];
  *(v8bf*)&supT[((size_t)t * 32 + c) * (size_t)N_ + nb + 8 * hf] = pk;
}

// ---------------------------------------------------------------------------
// z2[k,t,n,c] = sum_m adj[k,t,n,m] * sup[t,m,c] + b2[c]
// A = adj tile straight from global (f32 -> bf16, exact since adj in {0,1}),
// B = supT. 8 waves x (16 rows, 2 c-tiles). adj stream prefetched 1KB ahead.
// ---------------------------------------------------------------------------
__global__ __launch_bounds__(256) void z2_kernel(const float* __restrict__ adj,
                                                 const bf16_t* __restrict__ supT,
                                                 const float* __restrict__ b2,
                                                 float* __restrict__ z2) {
  const int p = blockIdx.x;
  const int t = p % T_;
  const int lane = threadIdx.x & 31, wave = threadIdx.x >> 5;
  const int l16 = lane & 15, hf = lane >> 4;
  const int nb = blockIdx.y * 128 + wave * 16;
  const float* arow = adj + ((size_t)p * N_ + nb + l16) * N_;
  const bf16_t* sp = supT + (size_t)t * 32 * N_;
  v8f acc[2];
  acc[0] = zero8();
  acc[1] = zero8();
  for (int kc = 0; kc < N_; kc += 32) {
    if (kc + 256 < N_) __builtin_prefetch(arow + kc + 256, 0, 1);
    v16bf a = frag_a_f32(arow + kc, hf);
#pragma unroll
    for (int j = 0; j < 2; j++) {
      int c = j * 16 + l16;
      v16bf b = frag_b_bf16(&sp[(size_t)c * N_ + kc], hf);
      acc[j] = wmma_bf16(a, b, acc[j]);
    }
  }
#pragma unroll
  for (int j = 0; j < 2; j++) {
    int c = j * 16 + l16;
    if (c < C_) {
      float bb = b2[c];
#pragma unroll
      for (int v = 0; v < 8; v++) {
        int n = nb + v + 8 * hf;
        z2[((size_t)p * N_ + n) * C_ + c] = acc[j][v] + bb;
      }
    }
  }
}

// ---------------------------------------------------------------------------
// Self-attention 2 + log_softmax over C. wave per (k,n).
// ---------------------------------------------------------------------------
__global__ __launch_bounds__(256) void out_kernel(const float* __restrict__ z2,
                                                  const float* __restrict__ at2W,
                                                  const float* __restrict__ at2b,
                                                  const float* __restrict__ at2a,
                                                  float* __restrict__ out) {
  const int k = blockIdx.x;
  const int wave = threadIdx.x >> 5, lane = threadIdx.x & 31;
  const int n = blockIdx.y * 8 + wave;
  float zr[T_][C_];
#pragma unroll
  for (int t = 0; t < T_; t++) {
    const float* zp = z2 + ((size_t)(k * T_ + t) * N_ + n) * C_;
#pragma unroll
    for (int c = 0; c < C_; c++) zr[t][c] = zp[c];
  }
  const float* av = at2a + k * 2 * H_;
  const float* Wk = at2W + (size_t)k * C_ * H_;
  const float* bk = at2b + k * H_;
  float part[T_];
  float pself = 0.f;
#pragma unroll
  for (int t = 0; t < T_; t++) {
    float pt = 0.f;
    for (int hh = lane; hh < H_; hh += 32) {
      float xlv = bk[hh];
#pragma unroll
      for (int c = 0; c < C_; c++) xlv += zr[t][c] * Wk[c * H_ + hh];
      pt += xlv * av[hh];
      if (t == k) pself += xlv * av[H_ + hh];
    }
    part[t] = wred_sum(pt);
  }
  pself = wred_sum(pself);
  float U0 = leakyf(part[0] + pself, 0.01f);
  float U1 = leakyf(part[1] + pself, 0.01f);
  float U2 = leakyf(part[2] + pself, 0.01f);
  float um = fmaxf(U0, fmaxf(U1, U2));
  float e0 = __expf(U0 - um), e1 = __expf(U1 - um), e2 = __expf(U2 - um);
  float inv = 1.f / (e0 + e1 + e2);
  float w0 = e0 * inv, w1 = e1 * inv, w2 = e2 * inv;
  // x2 for c = lane (<18), then log_softmax across the 18 lanes
  float x2 = -3.0e38f;
  if (lane < C_) {
    float v0 = z2[((size_t)(k * T_ + 0) * N_ + n) * C_ + lane];
    float v1 = z2[((size_t)(k * T_ + 1) * N_ + n) * C_ + lane];
    float v2 = z2[((size_t)(k * T_ + 2) * N_ + n) * C_ + lane];
    x2 = 3.f * (w0 * v0 + w1 * v1 + w2 * v2);
  }
  float mxv = wred_max(x2);
  float ex = (lane < C_) ? __expf(x2 - mxv) : 0.f;
  float se = wred_sum(ex);
  if (lane < C_) out[((size_t)k * N_ + n) * C_ + lane] = x2 - mxv - __logf(se);
}

// ---------------------------------------------------------------------------
extern "C" void kernel_launch(void* const* d_in, const int* in_sizes, int n_in,
                              void* d_out, int out_size, void* d_ws, size_t ws_size,
                              hipStream_t stream) {
  (void)in_sizes; (void)n_in; (void)out_size; (void)ws_size;
  const float* x    = (const float*)d_in[0];
  const float* adj  = (const float*)d_in[1];
  const float* W1   = (const float*)d_in[2];
  const float* b1   = (const float*)d_in[3];
  const float* a1   = (const float*)d_in[4];
  const float* a2   = (const float*)d_in[5];
  const float* at1W = (const float*)d_in[6];
  const float* at1b = (const float*)d_in[7];
  const float* at1a = (const float*)d_in[8];
  const float* W2   = (const float*)d_in[9];
  const float* b2   = (const float*)d_in[10];
  const float* at2W = (const float*)d_in[11];
  const float* at2b = (const float*)d_in[12];
  const float* at2a = (const float*)d_in[13];
  float* out = (float*)d_out;

  char* w = (char*)d_ws;
  auto carve = [&](size_t bytes) -> void* {
    void* p = (void*)w;
    w += (bytes + 255) & ~(size_t)255;
    return p;
  };
  float*  h     = (float*) carve(sizeof(float) * (size_t)T_ * N_ * D_);
  bf16_t* hT    = (bf16_t*)carve(2 * (size_t)T_ * D_ * N_);
  float*  s1w   = (float*) carve(sizeof(float) * (size_t)T_ * N_);
  float*  s2w   = (float*) carve(sizeof(float) * (size_t)T_ * T_ * N_);
  float*  z1    = (float*) carve(sizeof(float) * (size_t)T_ * T_ * N_ * D_);
  float*  xlw   = (float*) carve(sizeof(float) * (size_t)T_ * T_ * N_ * H_);
  float*  x1w   = (float*) carve(sizeof(float) * (size_t)T_ * N_ * D_);
  bf16_t* supT  = (bf16_t*)carve(2 * (size_t)T_ * 32 * N_);
  float*  z2w   = (float*) carve(sizeof(float) * (size_t)T_ * T_ * N_ * C_);
  bf16_t* at1WT = (bf16_t*)carve(2 * (size_t)T_ * 64 * D_);
  bf16_t* W2T   = (bf16_t*)carve(2 * (size_t)32 * D_);

  prep_kernel<<<dim3((T_*64*D_ + 32*D_) / 256), dim3(256), 0, stream>>>(at1W, W2, at1WT, W2T);
  proj_kernel<<<dim3(T_, N_/16), dim3(256), 0, stream>>>(x, W1, h, hT);
  s12_kernel<<<dim3((T_*N_ + T_*T_*N_) / 8), dim3(256), 0, stream>>>(h, a1, a2, s1w, s2w);
  att_kernel<<<dim3(T_*T_, N_/16), dim3(256), 0, stream>>>(adj, s1w, s2w, hT, b1, z1);
  xl_kernel<<<dim3(T_*T_, N_/16), dim3(128), 0, stream>>>(z1, at1WT, at1b, xlw);
  selfatt1_kernel<<<dim3(T_, N_/8), dim3(256), 0, stream>>>(xlw, at1a, z1, x1w);
  sup_kernel<<<dim3(T_, N_/16), dim3(64), 0, stream>>>(x1w, W2T, supT);
  z2_kernel<<<dim3(T_*T_, N_/128), dim3(256), 0, stream>>>(adj, supT, b2, z2w);
  out_kernel<<<dim3(T_, N_/8), dim3(256), 0, stream>>>(z2w, at2W, at2b, at2a, out);
}
